// AttnDecoder_68719477301
// MI455X (gfx1250) — matrix operand order repaired and checked by reading, not compiled, not run on hardware
//
#include <hip/hip_runtime.h>
#include <hip/hip_bf16.h>
#include <math.h>

// ---------------------------------------------------------------------------
// AttnDecoder for MI455X (gfx1250, wave32, WMMA).
//  B=256 S=512 T=48 IN=32 H=512 L=2, FC 512->256->...->8->1
// ---------------------------------------------------------------------------

typedef __attribute__((ext_vector_type(16))) __bf16 v16bf;
typedef __attribute__((ext_vector_type(8)))  float  v8f;

#define B_   256
#define S_   512
#define T_   48
#define IN_  32
#define H_   512
#define G3H  1536   // 3*H  (GEMM N dimension, 96 n-tiles of 16)
#define KMAX 544    // max GEMM K (ctx+x concat)

// Compile-safe probe for the gfx1250 async LDS-DMA path (ASYNCcnt).
// Round-2 diagnostic confirmed the builtin exists with param0 = v4i in AS1
// (HIP prints AS1 as `__device__`); use matching vector-pointee casts.
#if defined(__HIP_DEVICE_COMPILE__) && \
    __has_builtin(__builtin_amdgcn_global_load_async_to_lds_b128) && \
    __has_builtin(__builtin_amdgcn_s_wait_asynccnt)
#define USE_ASYNC_COPY 1
typedef int v4i __attribute__((ext_vector_type(4)));
typedef __attribute__((address_space(1))) v4i* as1_v4i_p;   // global
typedef __attribute__((address_space(3))) v4i* as3_v4i_p;   // LDS
#else
#define USE_ASYNC_COPY 0
#endif

// ---------------------------------------------------------------------------
// One-time: enc (S,B,H) f32  ->  (B,S,H) bf16  (halves attention HBM traffic)
// ---------------------------------------------------------------------------
__global__ __launch_bounds__(256) void enc_convert_k(const float* __restrict__ enc_in,
                                                     __bf16* __restrict__ enc_out) {
  size_t idx = (size_t)blockIdx.x * 256 + threadIdx.x;   // one thread = 4 h-elems
  int    hq  = (int)(idx & 127);                         // H/4 = 128
  size_t bs  = idx >> 7;
  int    s   = (int)(bs & (S_ - 1));
  int    b   = (int)(bs >> 9);
  float4 v = *(const float4*)(enc_in + (((size_t)s * B_ + b) * H_ + hq * 4));
  __bf16* o = enc_out + (((size_t)b * S_ + s) * H_ + hq * 4);
  o[0] = (__bf16)v.x; o[1] = (__bf16)v.y; o[2] = (__bf16)v.z; o[3] = (__bf16)v.w;
}

// ---------------------------------------------------------------------------
// One-time: W (3H x K) f32 row-major -> WMMA B-fragment-packed bf16.
// Tile (kt,nt) is 512 bf16, stored so lane L's 16 elements are contiguous:
//   n = nt*16 + L%16 ; k = kt*32 + 16*(L/16) + 2v + p   (contiguous k order, same as A in gemm_k)
// ---------------------------------------------------------------------------
__global__ __launch_bounds__(256) void pack_w_k(const float* __restrict__ W,
                                                __bf16* __restrict__ out,
                                                int K, int numKt) {
  int idx  = blockIdx.x * 256 + threadIdx.x;
  int tile = idx >> 9;
  if (tile >= numKt * (G3H / 16)) return;
  int r    = idx & 511;
  int lane = r >> 4;
  int e    = r & 15;
  int v    = e >> 1, pbit = e & 1;
  int nt   = tile / numKt;
  int kt   = tile - nt * numKt;
  int n    = nt * 16 + (lane & 15);
  int k    = kt * 32 + ((lane >> 4) << 4) + v * 2 + pbit;
  out[idx] = (__bf16)W[(size_t)n * K + k];
}

// ---------------------------------------------------------------------------
// One-time: hidden state init from encoder_state (L,B,H); pred feedback = 0
// ---------------------------------------------------------------------------
__global__ __launch_bounds__(256) void init_h_k(const float* __restrict__ es,
                                                float* h0f, float* h1f,
                                                __bf16* h0b, __bf16* h1b,
                                                float* pred) {
  int idx = blockIdx.x * 256 + threadIdx.x;
  if (idx < B_ * H_) {
    float a = es[idx], c = es[B_ * H_ + idx];
    h0f[idx] = a; h1f[idx] = c;
    h0b[idx] = (__bf16)a; h1b[idx] = (__bf16)c;
  }
  if (idx < B_) pred[idx] = 0.f;
}

// ---------------------------------------------------------------------------
// Fused single-pass attention (online softmax + running context).
// 1 WG per batch row, 8 waves; wave w handles s = w, w+8, ... (64 iters).
// Lane l owns h-range [16l, 16l+16).  enc is bf16 -> 6.1 GB total HBM,
// ~0.26 ms at 23.3 TB/s; loads forced to b128 via union casts.
// ---------------------------------------------------------------------------
__global__ __launch_bounds__(256) void attn_k(const float* __restrict__ h1f,
                                              const __bf16* __restrict__ encb,
                                              float* __restrict__ ctx) {
  __shared__ float sm[8], sl[8];
  __shared__ float sctx[8][H_];
  const int   b    = blockIdx.x;
  const int   lane = threadIdx.x & 31;
  const int   wv   = threadIdx.x >> 5;
  const float scale = 0.044194173824159216f;   // 1/sqrt(512)

  union { float4 f4[4]; float f[16]; } qu;
  const float4* qp = (const float4*)(h1f + (size_t)b * H_ + lane * 16);
#pragma unroll
  for (int j = 0; j < 4; ++j) qu.f4[j] = qp[j];

  float m = -INFINITY, l = 0.f, acc[16];
#pragma unroll
  for (int j = 0; j < 16; ++j) acc[j] = 0.f;

  const __bf16* ebase = encb + (size_t)b * S_ * H_ + lane * 16;
  for (int s = wv; s < S_; s += 8) {
    union { uint4 q[2]; __bf16 h[16]; } eu;
    const uint4* ep = (const uint4*)(ebase + (size_t)s * H_);
    eu.q[0] = ep[0];
    eu.q[1] = ep[1];
    float e[16];
#pragma unroll
    for (int j = 0; j < 16; ++j) e[j] = (float)eu.h[j];
    float part = 0.f;
#pragma unroll
    for (int j = 0; j < 16; ++j) part = fmaf(qu.f[j], e[j], part);
#pragma unroll
    for (int off = 16; off >= 1; off >>= 1) part += __shfl_xor(part, off, 32);
    float score = part * scale;
    float mn    = fmaxf(m, score);
    float alpha = __expf(m - mn);
    float p     = __expf(score - mn);
    l = l * alpha + p;
#pragma unroll
    for (int j = 0; j < 16; ++j) acc[j] = fmaf(acc[j], alpha, p * e[j]);
    m = mn;
  }

#pragma unroll
  for (int j = 0; j < 16; ++j) sctx[wv][lane * 16 + j] = acc[j];
  if (lane == 0) { sm[wv] = m; sl[wv] = l; }
  __syncthreads();

  float M = sm[0];
#pragma unroll
  for (int w = 1; w < 8; ++w) M = fmaxf(M, sm[w]);
  float w8[8], denom = 0.f;
#pragma unroll
  for (int w = 0; w < 8; ++w) { w8[w] = __expf(sm[w] - M); denom = fmaf(sl[w], w8[w], denom); }
  float inv = 1.f / denom;
  for (int hh = threadIdx.x; hh < H_; hh += 256) {
    float num = 0.f;
#pragma unroll
    for (int w = 0; w < 8; ++w) num = fmaf(sctx[w][hh], w8[w], num);
    ctx[(size_t)b * H_ + hh] = num * inv;
  }
}

// ---------------------------------------------------------------------------
// Build X0 = [ctx | x_t] bf16 (B x 544), with autoregressive feedback x[:, -1]
// ---------------------------------------------------------------------------
__global__ __launch_bounds__(256) void packx_k(const float* __restrict__ ctx,
                                               const float* __restrict__ inputs,
                                               const float* __restrict__ pred,
                                               __bf16* __restrict__ X0, int t) {
  int idx = blockIdx.x * 256 + threadIdx.x;
  if (idx >= B_ * (H_ + IN_)) return;
  int b = idx / (H_ + IN_);
  int c = idx - b * (H_ + IN_);
  float v;
  if (c < H_) v = ctx[(size_t)b * H_ + c];
  else {
    int j = c - H_;
    v = inputs[((size_t)b * T_ + t) * IN_ + j];
    if (j == IN_ - 1 && t > 0) v = pred[b];
  }
  X0[idx] = (__bf16)v;
}

// ---------------------------------------------------------------------------
// bf16 WMMA GEMM: C(256 x 1536) = A(256 x K) * Wpacked^T, f32 accumulate.
// Grid (16, 12) x 256 threads: WG owns a 16(M) x 128(N) strip; wave w owns
// n-tile blockIdx.y*8+w.  The whole 16 x K A-strip (<=17.4 KB of 320 KB LDS)
// is preloaded ONCE (via async LDS-DMA when the toolchain exposes it), so the
// K-loop is barrier-free: 2x ds_load_b128 + 2x global_load_b128 + 1x wmma.
// ---------------------------------------------------------------------------
__global__ __launch_bounds__(256) void gemm_k(const __bf16* __restrict__ A,
                                              const __bf16* __restrict__ Bp,
                                              float* __restrict__ C, int K) {
  __shared__ __align__(16) __bf16 sA[16 * KMAX];
  const int tid   = threadIdx.x;
  const int lane  = tid & 31;
  const int wv    = tid >> 5;
  const int mt    = blockIdx.x;
  const int nt    = blockIdx.y * 8 + wv;
  const int numKt = K >> 5;

  // ---- preload A strip: 16 rows x K bf16, 16-byte chunks ----
  const int cpr = K >> 3;                  // 16B chunks per row
  for (int c = tid; c < 16 * cpr; c += 256) {
    int row = c / cpr, col = c - row * cpr;
    const __bf16* g = A + (size_t)(mt * 16 + row) * K + col * 8;
    __bf16*       d = sA + (size_t)row * K + col * 8;
#if USE_ASYNC_COPY
    __builtin_amdgcn_global_load_async_to_lds_b128((as1_v4i_p)g, (as3_v4i_p)d, 0, 0);
#else
    *(uint4*)d = *(const uint4*)g;
#endif
  }
#if USE_ASYNC_COPY
  __builtin_amdgcn_s_wait_asynccnt(0);
#endif
  __syncthreads();

  // ---- barrier-free K loop ----
  const unsigned int* sAu = (const unsigned int*)sA;
  const int mloc = lane & 15;
  const int gsel = lane >> 4;
  const unsigned int* rowBase = sAu + (size_t)mloc * (K >> 1);
  const __bf16* bTile = Bp + ((size_t)nt * numKt) * 512 + lane * 16;

  v8f c = {};
  for (int kt = 0; kt < numKt; ++kt) {
    union { unsigned int u[8]; v16bf v; } afrag;
    const unsigned int* rp = rowBase + kt * 16;
#pragma unroll
    for (int vv = 0; vv < 8; ++vv) {
      int k0 = 2 * vv + 16 * gsel;   // contiguous k = 16*(lane/16) + s, matches packed B
      afrag.u[vv] = rp[k0 >> 1];
    }
    v16bf bfrag = *(const v16bf*)(bTile + (size_t)kt * 512);
    if (kt + 1 < numKt) __builtin_prefetch(bTile + (size_t)(kt + 1) * 512, 0, 1);
    c = __builtin_amdgcn_wmma_f32_16x16x32_bf16(false, afrag.v, false, bfrag,
                                                (short)0, c, false, false);
  }
#pragma unroll
  for (int r = 0; r < 8; ++r) {            // C layout: m = r + 8*(lane/16)
    int mrow = mt * 16 + r + 8 * gsel;
    C[(size_t)mrow * G3H + nt * 16 + (lane & 15)] = c[r];
  }
}

// ---------------------------------------------------------------------------
// GRU gate combine:  r=sig(ir+hr) z=sig(iz+hz) n=tanh(in + r*hn)
// h' = (1-z)*n + z*h ; writes f32 state + bf16 copy for the next GEMM.
// ---------------------------------------------------------------------------
__global__ __launch_bounds__(256) void gru_combine_k(const float* __restrict__ gi,
                                                     const float* __restrict__ gh,
                                                     const float* __restrict__ bih,
                                                     const float* __restrict__ bhh,
                                                     float* __restrict__ hf,
                                                     __bf16* __restrict__ hb) {
  int idx = blockIdx.x * 256 + threadIdx.x;     // B*H
  int b = idx >> 9;
  int h = idx & 511;
  const float* gib = gi + (size_t)b * G3H;
  const float* ghb = gh + (size_t)b * G3H;
  float ir  = gib[h]        + bih[h];
  float hr  = ghb[h]        + bhh[h];
  float iz  = gib[512 + h]  + bih[512 + h];
  float hz  = ghb[512 + h]  + bhh[512 + h];
  float in_ = gib[1024 + h] + bih[1024 + h];
  float hn  = ghb[1024 + h] + bhh[1024 + h];
  float r = 1.f / (1.f + __expf(-(ir + hr)));
  float z = 1.f / (1.f + __expf(-(iz + hz)));
  float n = tanhf(in_ + r * hn);
  float hnew = (1.f - z) * n + z * hf[idx];
  hf[idx] = hnew;
  hb[idx] = (__bf16)hnew;
}

// ---------------------------------------------------------------------------
// Fused 7-layer FC head, one WG per batch row, LDS ping-pong.
// Writes d_out[b][t] and the feedback buffer.
// ---------------------------------------------------------------------------
__global__ __launch_bounds__(256) void fc_k(const float* __restrict__ h1f,
    const float* W0, const float* W1, const float* W2, const float* W3,
    const float* W4, const float* W5, const float* W6,
    const float* c0, const float* c1, const float* c2, const float* c3,
    const float* c4, const float* c5, const float* c6,
    float* __restrict__ pred, float* __restrict__ out, int t) {
  __shared__ float bufA[512];
  __shared__ float bufB[256];
  const int b = blockIdx.x;
  const int tid = threadIdx.x;
  bufA[tid]       = h1f[(size_t)b * H_ + tid];
  bufA[tid + 256] = h1f[(size_t)b * H_ + tid + 256];
  __syncthreads();
  const float* Ws[7] = {W0, W1, W2, W3, W4, W5, W6};
  const float* bs[7] = {c0, c1, c2, c3, c4, c5, c6};
  const int dims[8]  = {512, 256, 128, 64, 32, 16, 8, 1};
  float* cur = bufA; float* nxt = bufB;
  for (int li = 0; li < 7; ++li) {
    int din = dims[li], dout = dims[li + 1];
    if (tid < dout) {
      const float* w = Ws[li] + (size_t)tid * din;
      float acc = bs[li][tid];
      for (int k = 0; k < din; ++k) acc = fmaf(w[k], cur[k], acc);
      if (li < 6) acc = fmaxf(acc, 0.f);
      nxt[tid] = acc;
    }
    __syncthreads();
    float* tmp = cur; cur = nxt; nxt = tmp;
  }
  if (tid == 0) { float pv = cur[0]; pred[b] = pv; out[(size_t)b * T_ + t] = pv; }
}

// ---------------------------------------------------------------------------
extern "C" void kernel_launch(void* const* d_in, const int* in_sizes, int n_in,
                              void* d_out, int out_size, void* d_ws, size_t ws_size,
                              hipStream_t stream) {
  const float* inputs    = (const float*)d_in[0];
  const float* enc_in    = (const float*)d_in[1];
  const float* enc_state = (const float*)d_in[2];
  const float* Wih0 = (const float*)d_in[3];
  const float* Wih1 = (const float*)d_in[4];
  const float* Whh0 = (const float*)d_in[5];
  const float* Whh1 = (const float*)d_in[6];
  const float* bih0 = (const float*)d_in[7];
  const float* bih1 = (const float*)d_in[8];
  const float* bhh0 = (const float*)d_in[9];
  const float* bhh1 = (const float*)d_in[10];
  const float* fcW[7]; const float* fcb[7];
  for (int i = 0; i < 7; ++i) { fcW[i] = (const float*)d_in[11 + i]; fcb[i] = (const float*)d_in[18 + i]; }
  float* out = (float*)d_out;

  // ---- carve workspace (~146 MB) ----
  char* p = (char*)d_ws;
  auto carve = [&](size_t bytes) -> void* {
    void* r = (void*)p; p += (bytes + 255) & ~(size_t)255; return r;
  };
  const int K0 = H_ + IN_, K1 = H_;
  const int nKt0 = K0 / 32, nKt1 = K1 / 32;
  __bf16* encb  = (__bf16*)carve((size_t)B_ * S_ * H_ * 2);
  __bf16* Wih0p = (__bf16*)carve((size_t)nKt0 * 96 * 512 * 2);
  __bf16* Whh0p = (__bf16*)carve((size_t)nKt1 * 96 * 512 * 2);
  __bf16* Wih1p = (__bf16*)carve((size_t)nKt1 * 96 * 512 * 2);
  __bf16* Whh1p = (__bf16*)carve((size_t)nKt1 * 96 * 512 * 2);
  float*  h0f   = (float*)carve((size_t)B_ * H_ * 4);
  float*  h1f   = (float*)carve((size_t)B_ * H_ * 4);
  __bf16* h0b   = (__bf16*)carve((size_t)B_ * H_ * 2);
  __bf16* h1b   = (__bf16*)carve((size_t)B_ * H_ * 2);
  float*  ctx   = (float*)carve((size_t)B_ * H_ * 4);
  __bf16* X0b   = (__bf16*)carve((size_t)B_ * K0 * 2);
  float*  gi    = (float*)carve((size_t)B_ * G3H * 4);
  float*  gh    = (float*)carve((size_t)B_ * G3H * 4);
  float*  pred  = (float*)carve((size_t)B_ * 4);

  // ---- one-time prep ----
  enc_convert_k<<<(B_ * S_ * H_ / 4 + 255) / 256, 256, 0, stream>>>(enc_in, encb);
  pack_w_k<<<(nKt0 * 96 * 512 + 255) / 256, 256, 0, stream>>>(Wih0, Wih0p, K0, nKt0);
  pack_w_k<<<(nKt1 * 96 * 512 + 255) / 256, 256, 0, stream>>>(Whh0, Whh0p, K1, nKt1);
  pack_w_k<<<(nKt1 * 96 * 512 + 255) / 256, 256, 0, stream>>>(Wih1, Wih1p, K1, nKt1);
  pack_w_k<<<(nKt1 * 96 * 512 + 255) / 256, 256, 0, stream>>>(Whh1, Whh1p, K1, nKt1);
  init_h_k<<<(B_ * H_ + 255) / 256, 256, 0, stream>>>(enc_state, h0f, h1f, h0b, h1b, pred);

  // ---- 48 sequential decode steps ----
  dim3 ggrid(16, 12);                     // 16 m-tiles x 12 n-strips of 8 tiles
  for (int t = 0; t < T_; ++t) {
    attn_k<<<B_, 256, 0, stream>>>(h1f, encb, ctx);
    packx_k<<<(B_ * K0 + 255) / 256, 256, 0, stream>>>(ctx, inputs, pred, X0b, t);
    gemm_k<<<ggrid, 256, 0, stream>>>(X0b, Wih0p, gi, K0);   // gi0
    gemm_k<<<ggrid, 256, 0, stream>>>(h0b, Whh0p, gh, K1);   // gh0 (h0 prev)
    gru_combine_k<<<(B_ * H_ + 255) / 256, 256, 0, stream>>>(gi, gh, bih0, bhh0, h0f, h0b);
    gemm_k<<<ggrid, 256, 0, stream>>>(h0b, Wih1p, gi, K1);   // gi1 (h0 new)
    gemm_k<<<ggrid, 256, 0, stream>>>(h1b, Whh1p, gh, K1);   // gh1 (h1 prev)
    gru_combine_k<<<(B_ * H_ + 255) / 256, 256, 0, stream>>>(gi, gh, bih1, bhh1, h1f, h1b);
    fc_k<<<B_, 256, 0, stream>>>(h1f,
        fcW[0], fcW[1], fcW[2], fcW[3], fcW[4], fcW[5], fcW[6],
        fcb[0], fcb[1], fcb[2], fcb[3], fcb[4], fcb[5], fcb[6], pred, out, t);
  }
  (void)in_sizes; (void)n_in; (void)out_size; (void)ws_size;
}
